// TCSLBPBlock_90357521973537
// MI455X (gfx1250) — matrix operand order repaired and checked by reading, not compiled, not run on hardware
//
#include <hip/hip_runtime.h>

#define C_INC 256
#define C_MIDC 512
#define HDIM 56
#define WDIM 56
#define BDIM 16
#define HW (HDIM*WDIM)          // 3136
#define NPIX (BDIM*HW)          // 50176

typedef __attribute__((ext_vector_type(16))) __bf16 v16bf;
typedef __attribute__((ext_vector_type(8)))  float  v8f;

__device__ __forceinline__ unsigned short f2bf(float f) {
  unsigned int u = __float_as_uint(f);
  u += 0x7FFFu + ((u >> 16) & 1u);        // round-to-nearest-even
  return (unsigned short)(u >> 16);
}

// ---------------------------------------------------------------------------
// Kernel 1: BN (inference) + NCHW fp32 -> NHWC bf16 transpose.
// ---------------------------------------------------------------------------
__global__ void bn_nhwc_kernel(const float* __restrict__ x,
                               const float* __restrict__ gamma,
                               const float* __restrict__ beta,
                               const float* __restrict__ rmean,
                               const float* __restrict__ rvar,
                               unsigned short* __restrict__ xbn) {
  __shared__ float ssc[C_INC];
  __shared__ float ssh[C_INC];
  __shared__ __align__(16) unsigned short t[WDIM * C_INC];   // 28,672 B

  const int bh = blockIdx.x;
  const int b = bh / HDIM;
  const int h = bh % HDIM;
  const int tid = threadIdx.x;

  {
    float sc = gamma[tid] * rsqrtf(rvar[tid] + 1e-5f);
    ssc[tid] = sc;
    ssh[tid] = beta[tid] - rmean[tid] * sc;
  }
  __syncthreads();

  for (int i = tid; i < C_INC * WDIM; i += 256) {
    int c = i / WDIM;
    int w = i % WDIM;
    float v = x[(((size_t)b * C_INC + c) * HDIM + h) * WDIM + w] * ssc[c] + ssh[c];
    t[w * C_INC + c] = f2bf(v);
  }
  __syncthreads();

  uint4* dst = (uint4*)(xbn + (size_t)bh * WDIM * C_INC);
  const uint4* src = (const uint4*)t;
  for (int i = tid; i < (WDIM * C_INC) / 8; i += 256) dst[i] = src[i];
}

// ---------------------------------------------------------------------------
// Kernel 2: repack w3 [512][256][3][3] fp32 -> bf16 A-fragments.
// A 16x32 layout (contiguous K): lanes 0-15 -> M=lane, K = j;
//                            lanes 16-31 -> M=lane-16, K = 16+j.
// ---------------------------------------------------------------------------
__global__ void repack_w3_kernel(const float* __restrict__ w3,
                                 unsigned short* __restrict__ w3r) {
  int o = blockIdx.x * 256 + threadIdx.x;   // total = 2304*512 = 1,179,648
  int within  = o & 511;
  int tileIdx = o >> 9;
  int lane = within >> 4;
  int j    = within & 15;
  int chunk  = tileIdx & 7;
  int tq     = tileIdx >> 3;     // octile*9 + tap
  int tap    = tq % 9;
  int octile = tq / 9;
  int m, k;
  if (lane < 16) { m = lane;      k = j;  }
  else           { m = lane - 16; k = j + 16; }
  int oc = octile * 16 + m;
  int ic = chunk * 32 + k;
  float v = w3[((size_t)oc * C_INC + ic) * 9 + tap];
  w3r[o] = f2bf(v);
}

// ---------------------------------------------------------------------------
// Kernel 3: repack w1 [256][512] fp32 -> bf16 A-fragments.
// ---------------------------------------------------------------------------
__global__ void repack_w1_kernel(const float* __restrict__ w1,
                                 unsigned short* __restrict__ w1r) {
  int o = blockIdx.x * 256 + threadIdx.x;   // total = 256*512 = 131,072
  int within  = o & 511;
  int tileIdx = o >> 9;
  int lane = within >> 4;
  int j    = within & 15;
  int chunk = tileIdx & 15;
  int ctile = tileIdx >> 4;
  int m, k;
  if (lane < 16) { m = lane;      k = j;  }
  else           { m = lane - 16; k = j + 16; }
  int c  = ctile * 16 + m;
  int kk = chunk * 32 + k;
  w1r[o] = f2bf(w1[(size_t)c * C_MIDC + kk]);
}

__device__ __forceinline__ void store_relu_bf16(v8f acc, unsigned short* dst) {
  unsigned int d[4];
#pragma unroll
  for (int r = 0; r < 4; ++r) {
    float f0 = acc[2 * r]     > 0.f ? acc[2 * r]     : 0.f;
    float f1 = acc[2 * r + 1] > 0.f ? acc[2 * r + 1] : 0.f;
    d[r] = (unsigned int)f2bf(f0) | ((unsigned int)f2bf(f1) << 16);
  }
  *(uint4*)dst = make_uint4(d[0], d[1], d[2], d[3]);
}

// ---------------------------------------------------------------------------
// Kernel 4: ternary 3x3 conv as 9 shifted GEMMs + ReLU -> mid (NHWC bf16).
// Tile: 16 wide x 2 h-rows per block (N=32). Halo 4x18x256 bf16 in LDS.
// Loop nest: chunk -> tap -> {2 LDS B loads (row0,row1); 4 A loads; 8 WMMAs}.
// A fragments reused by 8 WMMAs in registers; 8 live accumulators (64 VGPRs).
// ---------------------------------------------------------------------------
__global__ void conv3x3_kernel(const unsigned short* __restrict__ xbn,
                               const unsigned short* __restrict__ w3r,
                               unsigned short* __restrict__ mid) {
  __shared__ __align__(32) unsigned short tile[4 * 18 * C_INC];  // 36,864 B

  const int w0 = blockIdx.x * 16;
  const int h0 = blockIdx.y * 2;
  const int b  = blockIdx.z;
  const int tid  = threadIdx.x;
  const int lane = tid & 31;
  const int wave = tid >> 5;

  // Stage halo tile: rows h0-1..h0+2, cols w0-1..w0+16, all 256 ic (zero-pad OOB).
  for (int i = tid; i < 4 * 18 * 32; i += 256) {
    int pos = i >> 5;                  // 0..71  (row,col)
    int sub = i & 31;                  // 16-byte group within 256 channels
    int r = pos / 18, c = pos % 18;
    int hh = h0 - 1 + r, ww = w0 - 1 + c;
    uint4 v = make_uint4(0u, 0u, 0u, 0u);
    if (hh >= 0 && hh < HDIM && ww >= 0 && ww < WDIM) {
      v = *(const uint4*)(xbn + ((size_t)(b * HDIM + hh) * WDIM + ww) * C_INC + sub * 8);
    }
    *(uint4*)(tile + pos * C_INC + sub * 8) = v;
  }
  __syncthreads();

  const int n     = lane & 15;         // pixel column / B-matrix N
  const int khalf = lane >> 4;         // B: lanes16-31 hold K 16..31
  const bool valid = (w0 + n) < WDIM;

  // A fragment base for oc-group 0 (octile = wave); per-group stride = 576*512 halves.
  const unsigned short* abase = w3r + (size_t)(wave * 9) * 8 * 512 + lane * 16;
  const size_t gstride = (size_t)576 * 512;     // halves between oc-groups

  v8f z = {0.f,0.f,0.f,0.f,0.f,0.f,0.f,0.f};
  v8f acc0r0 = z, acc1r0 = z, acc2r0 = z, acc3r0 = z;   // output row h0
  v8f acc0r1 = z, acc1r1 = z, acc2r1 = z, acc3r1 = z;   // output row h0+1

  for (int chunk = 0; chunk < 8; ++chunk) {
    for (int tap = 0; tap < 9; ++tap) {
      const int dy = tap / 3, dx = tap % 3;
      const unsigned short* bbase = tile + (n + dx) * C_INC + chunk * 32 + khalf * 16;
      v16bf b0 = *(const v16bf*)(bbase + (dy)     * 18 * C_INC);   // input row h0-1+dy
      v16bf b1 = *(const v16bf*)(bbase + (dy + 1) * 18 * C_INC);   // input row h0+dy
      const unsigned short* ap = abase + (size_t)(tap * 8 + chunk) * 512;
      v16bf a0 = *(const v16bf*)(ap);
      v16bf a1 = *(const v16bf*)(ap + gstride);
      v16bf a2 = *(const v16bf*)(ap + 2 * gstride);
      v16bf a3 = *(const v16bf*)(ap + 3 * gstride);
      acc0r0 = __builtin_amdgcn_wmma_f32_16x16x32_bf16(false, a0, false, b0, (short)0, acc0r0, false, false);
      acc0r1 = __builtin_amdgcn_wmma_f32_16x16x32_bf16(false, a0, false, b1, (short)0, acc0r1, false, false);
      acc1r0 = __builtin_amdgcn_wmma_f32_16x16x32_bf16(false, a1, false, b0, (short)0, acc1r0, false, false);
      acc1r1 = __builtin_amdgcn_wmma_f32_16x16x32_bf16(false, a1, false, b1, (short)0, acc1r1, false, false);
      acc2r0 = __builtin_amdgcn_wmma_f32_16x16x32_bf16(false, a2, false, b0, (short)0, acc2r0, false, false);
      acc2r1 = __builtin_amdgcn_wmma_f32_16x16x32_bf16(false, a2, false, b1, (short)0, acc2r1, false, false);
      acc3r0 = __builtin_amdgcn_wmma_f32_16x16x32_bf16(false, a3, false, b0, (short)0, acc3r0, false, false);
      acc3r1 = __builtin_amdgcn_wmma_f32_16x16x32_bf16(false, a3, false, b1, (short)0, acc3r1, false, false);
    }
  }

  if (valid) {
    const int coff = wave * 16 + khalf * 8;
    unsigned short* d0 = mid + ((size_t)(b * HDIM + h0)     * WDIM + (w0 + n)) * C_MIDC + coff;
    unsigned short* d1 = mid + ((size_t)(b * HDIM + h0 + 1) * WDIM + (w0 + n)) * C_MIDC + coff;
    store_relu_bf16(acc0r0, d0);                 // octile = wave
    store_relu_bf16(acc1r0, d0 + 128 * 16);      // octile = wave + 8
    store_relu_bf16(acc2r0, d0 + 256 * 16);      // octile = wave + 16
    store_relu_bf16(acc3r0, d0 + 384 * 16);      // octile = wave + 24
    store_relu_bf16(acc0r1, d1);
    store_relu_bf16(acc1r1, d1 + 128 * 16);
    store_relu_bf16(acc2r1, d1 + 256 * 16);
    store_relu_bf16(acc3r1, d1 + 384 * 16);
  }
}

// ---------------------------------------------------------------------------
// Kernel 5: 1x1 conv (512->256) + bias + residual, NHWC bf16 mid -> NCHW fp32.
// 32 pixels per block; chunk -> {2 B loads; 2 A loads; 4 WMMAs}.
// ---------------------------------------------------------------------------
__global__ void conv1x1_kernel(const unsigned short* __restrict__ mid,
                               const unsigned short* __restrict__ w1r,
                               const float* __restrict__ b1,
                               const float* __restrict__ x,
                               float* __restrict__ out) {
  const int pixbase = blockIdx.x * 32;
  const int tid  = threadIdx.x;
  const int lane = tid & 31;
  const int wave = tid >> 5;
  const int n     = lane & 15;
  const int khalf = lane >> 4;

  const int p0 = pixbase + n;             // flat pixels, all valid (50176 % 32 == 0)
  const int p1 = p0 + 16;
  const size_t mid0 = (size_t)p0 * C_MIDC;
  const size_t mid1 = (size_t)p1 * C_MIDC;

  // A bases: ctile = wave (cg0) and wave+8 (cg1); per-cg stride = 8*16*512 halves.
  const unsigned short* abase = w1r + (size_t)(wave * 16) * 512 + lane * 16;
  const size_t cgstride = (size_t)8 * 16 * 512;

  v8f z = {0.f,0.f,0.f,0.f,0.f,0.f,0.f,0.f};
  v8f acc0p0 = z, acc0p1 = z, acc1p0 = z, acc1p1 = z;

#pragma unroll 2
  for (int chunk = 0; chunk < 16; ++chunk) {
    v16bf bm0 = *(const v16bf*)(mid + mid0 + chunk * 32 + khalf * 16);
    v16bf bm1 = *(const v16bf*)(mid + mid1 + chunk * 32 + khalf * 16);
    v16bf a0  = *(const v16bf*)(abase + (size_t)chunk * 512);
    v16bf a1  = *(const v16bf*)(abase + cgstride + (size_t)chunk * 512);
    acc0p0 = __builtin_amdgcn_wmma_f32_16x16x32_bf16(false, a0, false, bm0, (short)0, acc0p0, false, false);
    acc0p1 = __builtin_amdgcn_wmma_f32_16x16x32_bf16(false, a0, false, bm1, (short)0, acc0p1, false, false);
    acc1p0 = __builtin_amdgcn_wmma_f32_16x16x32_bf16(false, a1, false, bm0, (short)0, acc1p0, false, false);
    acc1p1 = __builtin_amdgcn_wmma_f32_16x16x32_bf16(false, a1, false, bm1, (short)0, acc1p1, false, false);
  }

#pragma unroll
  for (int cg = 0; cg < 2; ++cg) {
#pragma unroll
    for (int pt = 0; pt < 2; ++pt) {
      const v8f acc = (cg == 0) ? (pt == 0 ? acc0p0 : acc0p1)
                                : (pt == 0 ? acc1p0 : acc1p1);
      const int p = (pt == 0) ? p0 : p1;
      const int b = p / HW;
      const int rem = p % HW;
      const size_t xb = (size_t)b * C_INC * HW + rem;
      const int cbase = (cg * 8 + wave) * 16 + khalf * 8;
#pragma unroll
      for (int r = 0; r < 8; ++r) {
        const int c = cbase + r;
        const size_t idx = xb + (size_t)c * HW;
        out[idx] = acc[r] + b1[c] + x[idx];
      }
    }
  }
}

// ---------------------------------------------------------------------------
extern "C" void kernel_launch(void* const* d_in, const int* in_sizes, int n_in,
                              void* d_out, int out_size, void* d_ws, size_t ws_size,
                              hipStream_t stream) {
  (void)in_sizes; (void)n_in; (void)out_size; (void)ws_size;
  const float* x     = (const float*)d_in[0];
  const float* gamma = (const float*)d_in[1];
  const float* beta  = (const float*)d_in[2];
  const float* rmean = (const float*)d_in[3];
  const float* rvar  = (const float*)d_in[4];
  const float* w3    = (const float*)d_in[5];
  const float* w1    = (const float*)d_in[6];
  const float* b1    = (const float*)d_in[7];
  float* out = (float*)d_out;

  // Workspace layout (all 32B aligned):
  //   xbn 25,690,112 B | mid 51,380,224 B | w3r 2,359,296 B | w1r 262,144 B
  char* ws = (char*)d_ws;
  unsigned short* xbn = (unsigned short*)(ws);
  unsigned short* mid = (unsigned short*)(ws + 25690112ull);
  unsigned short* w3r = (unsigned short*)(ws + 25690112ull + 51380224ull);
  unsigned short* w1r = (unsigned short*)(ws + 25690112ull + 51380224ull + 2359296ull);

  bn_nhwc_kernel<<<BDIM * HDIM, 256, 0, stream>>>(x, gamma, beta, rmean, rvar, xbn);
  repack_w3_kernel<<<4608, 256, 0, stream>>>(w3, w3r);
  repack_w1_kernel<<<512, 256, 0, stream>>>(w1, w1r);
  conv3x3_kernel<<<dim3(4, HDIM / 2, BDIM), 256, 0, stream>>>(xbn, w3r, mid);
  conv1x1_kernel<<<NPIX / 32, 256, 0, stream>>>(mid, w1r, b1, x, out);
}
